// SimpleLSTM_65609920413978
// MI455X (gfx1250) — matrix operand order repaired and checked by reading, not compiled, not run on hardware
//
#include <hip/hip_runtime.h>
#include <hip/hip_bf16.h>
#include <math.h>

typedef __attribute__((ext_vector_type(16))) _Float16 v16h;
typedef __attribute__((ext_vector_type(8)))  float    v8f;

#define Hc   512
#define G4H  2048   // 4*H
#define Bc   512

union V16 { v16h v; float4 f4[2]; };

__device__ __forceinline__ float sigm(float x) { return 1.0f / (1.0f + expf(-x)); }

__device__ __forceinline__ v8f wmma_f16(const V16& a, const V16& b, v8f c) {
    return __builtin_amdgcn_wmma_f32_16x16x32_f16(
        false, a.v, false, b.v, (short)0, c, false, false);
}

// ---------------------------------------------------------------------------
// Convert W_hh [2048 x 512] f32 -> f16 (row-major, same layout)
// ---------------------------------------------------------------------------
__global__ __launch_bounds__(256) void cvt_w_kernel(const float* __restrict__ w,
                                                    _Float16* __restrict__ w16) {
    int i = blockIdx.x * 256 + threadIdx.x;
    w16[i] = (_Float16)w[i];
}

// ---------------------------------------------------------------------------
// Zero-init h16 / c  (B*H elements)
// ---------------------------------------------------------------------------
__global__ __launch_bounds__(256) void zero_kernel(_Float16* __restrict__ h16,
                                                   float* __restrict__ c) {
    int i = blockIdx.x * 256 + threadIdx.x;
    h16[i] = (_Float16)0.0f;
    c[i]   = 0.0f;
}

// ---------------------------------------------------------------------------
// FUSED LSTM step. One wave owns a 32x16 hidden slab (2 batch M-tiles x one
// hidden-col N-tile) and 8 WMMA accumulators (4 gates x 2 M-tiles). Each
// gate B-tile load feeds 2 WMMAs; the shared A tiles feed 4 each. K loop is
// fully unrolled so loads multi-buffer ahead of the WMMAs. Epilogue applies
// x_t*W_ih + bias, sigmoid/tanh cell update, writes c / h16(next) /
// hf(final step). hin/hout ping-pong to avoid cross-block RAW.
// grid = (4, 16), block = 256 (8 waves):
//   wave -> nTile = (bx*8+w)*16 (0..511), mTiles = by*32 and by*32+16
// ---------------------------------------------------------------------------
__global__ __launch_bounds__(256) void lstm_step_kernel(
    const _Float16* __restrict__ hin,   // [B, H] f16
    const _Float16* __restrict__ w16,   // [4H, H] f16
    const float* __restrict__ x,        // [B, T]
    const float* __restrict__ W_ih,     // [4H]
    const float* __restrict__ b_ih,     // [4H]
    const float* __restrict__ b_hh,     // [4H]
    float* __restrict__ c,              // [B, H] in/out
    float* __restrict__ hf,             // [B, H] f32 (written on last step)
    _Float16* __restrict__ hout,        // [B, H] f16 out
    int t, int T)
{
    const int lane  = threadIdx.x & 31;
    const int wave  = threadIdx.x >> 5;
    const int nTile = (blockIdx.x * 8 + wave) * 16;   // hidden col tile (0..511)
    const int mTile = blockIdx.y * 32;                // batch row slab (2 tiles)
    const int half  = lane >> 4;
    const int r     = lane & 15;

    const _Float16* aRow0 = hin + (mTile + r) * Hc;
    const _Float16* aRow1 = hin + (mTile + 16 + r) * Hc;
    const _Float16* bRowI = w16 + (nTile + r) * Hc;             // gate i
    const _Float16* bRowF = w16 + (Hc   + nTile + r) * Hc;      // gate f
    const _Float16* bRowG = w16 + (2*Hc + nTile + r) * Hc;      // gate g
    const _Float16* bRowO = w16 + (3*Hc + nTile + r) * Hc;      // gate o

    // acc[gate][mt]
    v8f aI0 = {}, aI1 = {}, aF0 = {}, aF1 = {};
    v8f aG0 = {}, aG1 = {}, aO0 = {}, aO1 = {};

    #pragma unroll
    for (int kb = 0; kb < Hc; kb += 32) {
        // A 16x32 f16: lane half 0 -> K = kb+0..7, kb+16..23; half 1 -> +8
        const int ka0 = kb + half * 8;
        const int ka1 = kb + 16 + half * 8;
        // B 32x16 f16: lane half 0 -> K = kb+0..7, kb+16..23; half 1 -> +8
        const int kb0 = kb + half * 8;

        V16 a0, a1, bI, bF, bG, bO;
        a0.f4[0] = *(const float4*)(aRow0 + ka0);
        a0.f4[1] = *(const float4*)(aRow0 + ka1);
        a1.f4[0] = *(const float4*)(aRow1 + ka0);
        a1.f4[1] = *(const float4*)(aRow1 + ka1);
        bI.f4[0] = *(const float4*)(bRowI + kb0);
        bI.f4[1] = *(const float4*)(bRowI + kb0 + 16);
        bF.f4[0] = *(const float4*)(bRowF + kb0);
        bF.f4[1] = *(const float4*)(bRowF + kb0 + 16);
        bG.f4[0] = *(const float4*)(bRowG + kb0);
        bG.f4[1] = *(const float4*)(bRowG + kb0 + 16);
        bO.f4[0] = *(const float4*)(bRowO + kb0);
        bO.f4[1] = *(const float4*)(bRowO + kb0 + 16);

        aI0 = wmma_f16(a0, bI, aI0);  aI1 = wmma_f16(a1, bI, aI1);
        aF0 = wmma_f16(a0, bF, aF0);  aF1 = wmma_f16(a1, bF, aF1);
        aG0 = wmma_f16(a0, bG, aG0);  aG1 = wmma_f16(a1, bG, aG1);
        aO0 = wmma_f16(a0, bO, aO0);  aO1 = wmma_f16(a1, bO, aO1);
    }

    // Epilogue. C/D 16x16 f32 layout: lane l holds N = l&15, M = (l>>4)*8+v.
    const int n = nTile + r;
    const float wiI = W_ih[n],        bI_ = b_ih[n]        + b_hh[n];
    const float wiF = W_ih[Hc + n],   bF_ = b_ih[Hc + n]   + b_hh[Hc + n];
    const float wiG = W_ih[2*Hc + n], bG_ = b_ih[2*Hc + n] + b_hh[2*Hc + n];
    const float wiO = W_ih[3*Hc + n], bO_ = b_ih[3*Hc + n] + b_hh[3*Hc + n];

    #pragma unroll
    for (int mt = 0; mt < 2; ++mt) {
        const v8f& vI = mt ? aI1 : aI0;
        const v8f& vF = mt ? aF1 : aF0;
        const v8f& vG = mt ? aG1 : aG0;
        const v8f& vO = mt ? aO1 : aO0;
        #pragma unroll
        for (int v = 0; v < 8; ++v) {
            const int m   = mTile + mt * 16 + half * 8 + v;
            const int idx = m * Hc + n;
            const float xv = x[m * T + t];     // x[b, t, 0]
            const float gi = vI[v] + xv * wiI + bI_;
            const float gf = vF[v] + xv * wiF + bF_;
            const float gg = vG[v] + xv * wiG + bG_;
            const float go = vO[v] + xv * wiO + bO_;
            const float cn = sigm(gf) * c[idx] + sigm(gi) * tanhf(gg);
            const float hn = sigm(go) * tanhf(cn);
            c[idx]    = cn;
            hout[idx] = (_Float16)hn;
            if (t == T - 1) hf[idx] = hn;      // decoder needs f32 hidden
        }
    }
}

// ---------------------------------------------------------------------------
// Plain GEMM (raw gates, no bias): gFix[b, n] = sum_k h16[b,k] * W16[n,k]
// Used once to precompute the FIXED decoder term hidden @ W_hh.T.
// grid = (16, 32), block = 256 (8 waves), one 16x16 tile per wave.
// ---------------------------------------------------------------------------
__global__ __launch_bounds__(256) void lstm_gemm_kernel(const _Float16* __restrict__ h16,
                                                        const _Float16* __restrict__ w16,
                                                        float* __restrict__ gates) {
    const int lane  = threadIdx.x & 31;
    const int wave  = threadIdx.x >> 5;
    const int nTile = (blockIdx.x * 8 + wave) * 16;
    const int mTile = blockIdx.y * 16;
    const int half  = lane >> 4;
    const int r     = lane & 15;

    const _Float16* aRow = h16 + (mTile + r) * Hc;
    const _Float16* bRow = w16 + (nTile + r) * Hc;

    v8f acc = {};
    #pragma unroll
    for (int kb = 0; kb < Hc; kb += 32) {
        const int ka0 = kb + half * 8;
        const int ka1 = kb + 16 + half * 8;
        const int kb0 = kb + half * 8;
        V16 a, b;
        a.f4[0] = *(const float4*)(aRow + ka0);
        a.f4[1] = *(const float4*)(aRow + ka1);
        b.f4[0] = *(const float4*)(bRow + kb0);
        b.f4[1] = *(const float4*)(bRow + kb0 + 16);
        acc = wmma_f16(a, b, acc);
    }

    const int n = nTile + r;
    #pragma unroll
    for (int v = 0; v < 8; ++v) {
        const int m = mTile + half * 8 + v;
        gates[m * G4H + n] = acc[v];
    }
}

// ---------------------------------------------------------------------------
// Decoder: hidden/cell are FIXED each step, so gFix = hidden @ W_hh.T is
// precomputed. One wave per batch element; h_t lives in 16 regs per lane
// (j = lane + 32*u). All TL steps run inside the kernel.
// ---------------------------------------------------------------------------
__global__ __launch_bounds__(256) void decoder_kernel(const float* __restrict__ gFix,
                                                      const float* __restrict__ hf,
                                                      const float* __restrict__ c,
                                                      const float* __restrict__ W_ih,
                                                      const float* __restrict__ b_ih,
                                                      const float* __restrict__ b_hh,
                                                      const float* __restrict__ W_out,
                                                      const float* __restrict__ b_out,
                                                      float* __restrict__ out,
                                                      int TL) {
    const int lane = threadIdx.x & 31;
    const int wave = threadIdx.x >> 5;
    const int b    = blockIdx.x * 8 + wave;

    float h[16], wout[16], cell[16];
    #pragma unroll
    for (int u = 0; u < 16; ++u) {
        const int j = lane + 32 * u;
        h[u]    = hf[b * Hc + j];
        wout[u] = W_out[j];
        cell[u] = c[b * Hc + j];
    }
    const float bo = b_out[0];

    for (int s = 0; s < TL; ++s) {
        // out = h_t . W_out + b_out  (wave32 shuffle reduction)
        float p = 0.0f;
        #pragma unroll
        for (int u = 0; u < 16; ++u) p += h[u] * wout[u];
        #pragma unroll
        for (int off = 16; off >= 1; off >>= 1) p += __shfl_xor(p, off, 32);
        const float o = p + bo;
        if (lane == 0) out[b * TL + s] = o;

        // h_new = lstm_cell(out, hidden, cell) with hidden@W_hh.T precomputed
        #pragma unroll
        for (int u = 0; u < 16; ++u) {
            const int j    = lane + 32 * u;
            const int base = b * G4H;
            const float gi = gFix[base + j]        + o * W_ih[j]        + b_ih[j]        + b_hh[j];
            const float gf = gFix[base + Hc + j]   + o * W_ih[Hc + j]   + b_ih[Hc + j]   + b_hh[Hc + j];
            const float gg = gFix[base + 2*Hc + j] + o * W_ih[2*Hc + j] + b_ih[2*Hc + j] + b_hh[2*Hc + j];
            const float go = gFix[base + 3*Hc + j] + o * W_ih[3*Hc + j] + b_ih[3*Hc + j] + b_hh[3*Hc + j];
            const float cn = sigm(gf) * cell[u] + sigm(gi) * tanhf(gg);
            h[u] = sigm(go) * tanhf(cn);
        }
    }
}

// ---------------------------------------------------------------------------
extern "C" void kernel_launch(void* const* d_in, const int* in_sizes, int n_in,
                              void* d_out, int out_size, void* d_ws, size_t ws_size,
                              hipStream_t stream) {
    const float* x     = (const float*)d_in[0];
    const float* W_ih  = (const float*)d_in[1];   // [2048, 1] -> vector of 2048
    const float* W_hh  = (const float*)d_in[2];   // [2048, 512]
    const float* b_ih  = (const float*)d_in[3];
    const float* b_hh  = (const float*)d_in[4];
    const float* W_out = (const float*)d_in[5];   // [1, 512]
    const float* b_out = (const float*)d_in[6];
    float* out = (float*)d_out;

    const int T  = in_sizes[0] / Bc;   // 256
    const int TL = out_size / Bc;      // 42 (task_level)

    // Workspace layout (offsets in bytes)
    char* ws = (char*)d_ws;
    _Float16* W16 = (_Float16*)(ws);              // 2048*512*2 = 2 MB
    _Float16* hA  = (_Float16*)(ws + 2097152);    // 512*512*2
    _Float16* hB  = (_Float16*)(ws + 2621440);    // 512*512*2
    float*    c   = (float*)(ws + 3145728);       // 512*512*4
    float*    hf  = (float*)(ws + 4194304);       // 512*512*4
    float*    gFix= (float*)(ws + 5242880);       // 512*2048*4
    (void)ws_size; (void)n_in;

    cvt_w_kernel<<<(G4H * Hc) / 256, 256, 0, stream>>>(W_hh, W16);
    zero_kernel<<<(Bc * Hc) / 256, 256, 0, stream>>>(hA, c);

    // Encoder recurrence: ONE fused WMMA+cell kernel per step, h ping-pong
    const dim3 sgrid(4, 16);   // (512/16)/8 waves in N, 512/32 slabs in M
    const _Float16* cur = hA;
    _Float16*       nxt = hB;
    for (int t = 0; t < T; ++t) {
        lstm_step_kernel<<<sgrid, 256, 0, stream>>>(
            cur, W16, x, W_ih, b_ih, b_hh, c, hf, nxt, t, T);
        const _Float16* tmp = cur; cur = nxt; nxt = (_Float16*)tmp;
    }

    // Fixed decoder gates: hidden @ W_hh.T (hidden/cell never change in decode)
    lstm_gemm_kernel<<<dim3(16, 32), 256, 0, stream>>>(cur, W16, gFix);

    // All TL decode steps in one kernel, one wave per batch element
    decoder_kernel<<<Bc / 8, 256, 0, stream>>>(
        gFix, hf, c, W_ih, b_ih, b_hh, W_out, b_out, out, TL);
}
